// Head_7215545057619
// MI455X (gfx1250) — matrix operand order repaired and checked by reading, not compiled, not run on hardware
//
#include <hip/hip_runtime.h>
#include <hip/hip_bf16.h>

#define B_  8
#define T_  2048
#define C_  1024
#define HS  64

typedef __bf16 bfx16 __attribute__((ext_vector_type(16)));
typedef __bf16 bfx8  __attribute__((ext_vector_type(8)));
typedef float  f32x8 __attribute__((ext_vector_type(8)));
typedef unsigned int u32x4 __attribute__((ext_vector_type(4)));
typedef int          i32x4 __attribute__((ext_vector_type(4)));
typedef int          i32x8 __attribute__((ext_vector_type(8)));

// Weight slice staged in LDS each K-step: 192 rows (3 weights x 64 heads) x 32 bf16.
// TDM pads each 64B row with 16B -> 80B row stride (16-bank spread for B-tile reads).
#define W_SLICE_ROWS 192
#define W_ROW_ELEMS  40                       // 32 payload + 8 pad bf16
#define W_SLICE_ELEMS (W_SLICE_ROWS * W_ROW_ELEMS)   // 7680 bf16 = 15360 B per buffer

#if defined(__gfx1250__) && __has_builtin(__builtin_amdgcn_tensor_load_to_lds) && (__clang_major__ < 23)
#define USE_TDM 1                              // ROCm 7.2 probe-verified 5-arg TDM builtin
#else
#define USE_TDM 0
#endif

#if defined(__gfx1250__) && __has_builtin(__builtin_amdgcn_sched_group_barrier)
#define SGB(mask, size, id) __builtin_amdgcn_sched_group_barrier((mask), (size), (id))
#else
#define SGB(mask, size, id)
#endif

// -------- wave32 cross-lane helpers (explicit ds_bpermute) -------------------------------
__device__ __forceinline__ float lane_xor_f(float v, int m) {
    int idx = (((int)(threadIdx.x & 31)) ^ m) << 2;
    return __int_as_float(__builtin_amdgcn_ds_bpermute(idx, __float_as_int(v)));
}
__device__ __forceinline__ float rmax16(float v) {
    v = fmaxf(v, lane_xor_f(v, 1)); v = fmaxf(v, lane_xor_f(v, 2));
    v = fmaxf(v, lane_xor_f(v, 4)); v = fmaxf(v, lane_xor_f(v, 8));
    return v;
}
__device__ __forceinline__ float rsum16(float v) {
    v += lane_xor_f(v, 1); v += lane_xor_f(v, 2);
    v += lane_xor_f(v, 4); v += lane_xor_f(v, 8);
    return v;
}
__device__ __forceinline__ bfx16 load_bf16x16(const __bf16* p) {
    bfx8 a = *(const bfx8*)p;
    bfx8 b = *(const bfx8*)(p + 8);
    bfx16 r;
#pragma unroll
    for (int i = 0; i < 8; i++) { r[i] = a[i]; r[i + 8] = b[i]; }
    return r;
}

// Issue one weight K-slice copy: gsrc = wbf + k0 (rows strided C_ elements) -> LDS buffer.
__device__ __forceinline__ void w_slice_issue(__bf16* lds_base, const __bf16* gsrc, int tid) {
#if USE_TDM
    if (tid < 32) {                            // one wave issues the DMA (EXEC-independent)
        unsigned long long ga = (unsigned long long)(size_t)gsrc;   // byte address
        unsigned la = (unsigned)(size_t)lds_base;                   // LDS byte offset
        u32x4 g0;
        g0.x = 1u;                                                  // count=1 (valid), user mode
        g0.y = la;                                                  // lds_addr
        g0.z = (unsigned)(ga & 0xffffffffull);                      // global_addr[31:0]
        g0.w = (unsigned)((ga >> 32) & 0x1ffffffull) | (2u << 30);  // addr[56:32] | type=2
        i32x8 g1;
        g1[0] = (1 << 16) | (1 << 20) | (3 << 22) | (3 << 25);      // data_size=2B, pad_en,
                                                                    // interval=16DW, amount=4DW
        g1[1] = (int)((C_ & 0xffff) << 16);                         // tensor_dim0 = 1024 (lo16)
        g1[2] = (int)(W_SLICE_ROWS << 16);                          // dim0 hi=0 | tensor_dim1=192
        g1[3] = (int)(32 << 16);                                    // dim1 hi=0 | tile_dim0=32
        g1[4] = W_SLICE_ROWS;                                       // tile_dim1=192, tile_dim2=0
        g1[5] = C_;                                                 // tensor_dim0_stride = 1024
        g1[6] = 0;
        g1[7] = 0;
        i32x4 gz = {0, 0, 0, 0};                                    // 2D: groups 2/3 unused
        __builtin_amdgcn_tensor_load_to_lds(g0, g1, gz, gz, 0);
    }
#else
    // fallback: cooperative regular loads + LDS stores (768 x 16B chunks, 128 threads)
#pragma unroll
    for (int j = 0; j < 6; j++) {
        int c = tid + 128 * j;
        int row = c >> 2, part = c & 3;
        *(bfx8*)(lds_base + row * W_ROW_ELEMS + part * 8) =
            *(const bfx8*)(gsrc + (size_t)row * C_ + part * 8);
    }
#endif
}

// -------- kernel 0: weights f32 -> bf16 --------------------------------------------------
__global__ __launch_bounds__(256) void convert_w_kernel(
    const float* __restrict__ Wq, const float* __restrict__ Wk,
    const float* __restrict__ Wv, __bf16* __restrict__ wbf)
{
    int i = blockIdx.x * 256 + threadIdx.x;
    wbf[i]               = (__bf16)Wq[i];
    wbf[HS * C_ + i]     = (__bf16)Wk[i];
    wbf[2 * HS * C_ + i] = (__bf16)Wv[i];
}

// -------- kernel 1: fused QKV projection + RoPE, TDM-staged weights ----------------------
__global__ __launch_bounds__(128) void qkv_rope_kernel(
    const float* __restrict__ x, const __bf16* __restrict__ wbf,
    float* __restrict__ out_k, float* __restrict__ out_v,
    __bf16* __restrict__ qbf, __bf16* __restrict__ kbf, __bf16* __restrict__ vTbf)
{
    __shared__ __align__(16) __bf16 wsl[2][W_SLICE_ELEMS];  // double-buffered, 30 KB

    const int tid  = threadIdx.x;
    const int lane = tid & 31;
    const int li = lane & 15, hi = lane >> 4;
    const int tile = blockIdx.x * 4 + (tid >> 5);
    const int g0 = tile * 16;
    const int b  = g0 / T_;
    const int t0 = g0 % T_;
    const float* xrow = x + (size_t)(g0 + li) * C_;

    f32x8 acc[12] = {};

    w_slice_issue(&wsl[0][0], wbf, tid);                    // prologue: slice k0=0

    for (int k0 = 0; k0 < C_; k0 += 32) {
        const int  cur = (k0 >> 5) & 1;
        const bool have_next = (k0 + 32) < C_;
        if (have_next) w_slice_issue(&wsl[cur ^ 1][0], wbf + k0 + 32, tid);
#if USE_TDM
        if (tid < 32) {                                     // wait current slice complete
            if (have_next) __builtin_amdgcn_s_wait_tensorcnt(1);
            else           __builtin_amdgcn_s_wait_tensorcnt(0);
        }
#endif
        __syncthreads();

        if (k0 + 32 < C_) __builtin_prefetch(xrow + k0 + 32 + hi * 8, 0, 3);
        // A-matrix 16x32 bf16: lane<16 -> K{0..15}, lane>=16 -> K{16..31}
        const float4 u0 = *(const float4*)(xrow + k0 + hi * 16);
        const float4 u1 = *(const float4*)(xrow + k0 + hi * 16 + 4);
        const float4 u2 = *(const float4*)(xrow + k0 + hi * 16 + 8);
        const float4 u3 = *(const float4*)(xrow + k0 + hi * 16 + 12);
        bfx16 a;
        a[0]=(__bf16)u0.x;  a[1]=(__bf16)u0.y;  a[2]=(__bf16)u0.z;  a[3]=(__bf16)u0.w;
        a[4]=(__bf16)u1.x;  a[5]=(__bf16)u1.y;  a[6]=(__bf16)u1.z;  a[7]=(__bf16)u1.w;
        a[8]=(__bf16)u2.x;  a[9]=(__bf16)u2.y;  a[10]=(__bf16)u2.z; a[11]=(__bf16)u2.w;
        a[12]=(__bf16)u3.x; a[13]=(__bf16)u3.y; a[14]=(__bf16)u3.z; a[15]=(__bf16)u3.w;

        // B-matrix 32x16 reads from LDS: row = w*64 + 16n + li, 16 contiguous bf16 at col hi*16.
        const __bf16* pb = &wsl[cur][0] + li * W_ROW_ELEMS + hi * 16;
#pragma unroll
        for (int t = 0; t < 12; t++) {
            bfx16 bm = load_bf16x16(pb + t * 16 * W_ROW_ELEMS);
            acc[t] = __builtin_amdgcn_wmma_f32_16x16x32_bf16(
                false, a, false, bm, (short)0, acc[t], false, false);
        }
        // Scheduler pipeline spec for this region (bounded by the two barriers):
        //   A-tile VMEM reads first, 3 B tiles (6 DS reads) of lookahead,
        //   then steady-state 1 WMMA : 2 DS reads, drain with 3 WMMAs.
        SGB(0x020, 4, 0);                     // 4 x VMEM read  (x A-tile)
        SGB(0x100, 6, 0);                     // 6 x DS read    (B tiles 0..2)
#pragma unroll
        for (int t = 0; t < 9; t++) {
            SGB(0x008, 1, 0);                 // 1 x WMMA
            SGB(0x100, 2, 0);                 // 2 x DS read (next B tile)
        }
        SGB(0x008, 3, 0);                     // drain: last 3 WMMAs
        __syncthreads();                      // readers done before overwrite
    }

    // RoPE on q,k (pair partner h^1 sits in adjacent lane of C layout) + stores
#pragma unroll
    for (int n = 0; n < 4; n++) {
        const int h  = 16 * n + li;
        const int pi = h >> 1;
        const float inv = __expf(-0.28782313662425572f * (float)pi); // 10000^(-2*pi/64)
#pragma unroll
        for (int r = 0; r < 8; r++) {
            const int m = r + 8 * hi;
            const int t = t0 + m;
            const int g = g0 + m;
            const float ang = (float)t * inv;
            const float cs = __cosf(ang), sn = __sinf(ang);
            float qv = acc[n][r];
            float qp = lane_xor_f(qv, 1);
            float qo = (h & 1) ? (qp * sn + qv * cs) : (qv * cs - qp * sn);
            float kv = acc[4 + n][r];
            float kp = lane_xor_f(kv, 1);
            float ko = (h & 1) ? (kp * sn + kv * cs) : (kv * cs - kp * sn);
            float vv = acc[8 + n][r];
            const size_t oidx = (size_t)g * HS + h;
            out_k[oidx] = ko;
            out_v[oidx] = vv;
            qbf[oidx]   = (__bf16)qo;
            kbf[oidx]   = (__bf16)ko;
            vTbf[(size_t)b * HS * T_ + (size_t)h * T_ + t] = (__bf16)vv;
        }
    }
}

// -------- kernel 2: causal flash attention, one wave per 16-query tile -------------------
__global__ __launch_bounds__(32) void attn_kernel(
    const __bf16* __restrict__ qbf, const __bf16* __restrict__ kbf,
    const __bf16* __restrict__ vTbf, float* __restrict__ out)
{
    __shared__ __align__(16) __bf16 pbuf[16 * 32];
    const int lane = threadIdx.x & 31;
    const int li = lane & 15, hi = lane >> 4;
    const int m0 = blockIdx.x * 16;
    const int b  = blockIdx.y;
    const __bf16* qb = qbf  + (size_t)b * T_ * HS;
    const __bf16* kb = kbf  + (size_t)b * T_ * HS;
    const __bf16* vb = vTbf + (size_t)b * HS * T_;

    bfx16 aq[2];
#pragma unroll
    for (int ks = 0; ks < 2; ks++) {
        const __bf16* p = qb + (size_t)(m0 + li) * HS + ks * 32 + hi * 16;
        bfx8 c0 = *(const bfx8*)p;
        bfx8 c1 = *(const bfx8*)(p + 8);
#pragma unroll
        for (int i = 0; i < 8; i++) { aq[ks][i] = c0[i]; aq[ks][8 + i] = c1[i]; }
    }

    f32x8 o[4] = {};
    float mrow[8], lrow[8];
#pragma unroll
    for (int r = 0; r < 8; r++) { mrow[r] = -1e30f; lrow[r] = 0.f; }

    const int kend = m0 + 16;
    for (int c0 = 0; c0 < kend; c0 += 32) {
        // S = q @ k^T : two 16x16 C-tiles, K=64 via 2 WMMAs each
        f32x8 s[2];
#pragma unroll
        for (int n = 0; n < 2; n++) {
            bfx16 bk0 = load_bf16x16(kb + (size_t)(c0 + 16 * n + li) * HS + hi * 16);
            bfx16 bk1 = load_bf16x16(kb + (size_t)(c0 + 16 * n + li) * HS + 32 + hi * 16);
            f32x8 cz = {};
            cz = __builtin_amdgcn_wmma_f32_16x16x32_bf16(false, aq[0], false, bk0, (short)0, cz, false, false);
            cz = __builtin_amdgcn_wmma_f32_16x16x32_bf16(false, aq[1], false, bk1, (short)0, cz, false, false);
            s[n] = cz;
        }
        // prefetch V tiles now so their latency hides behind the softmax VALU work
        bfx16 bv[4];
#pragma unroll
        for (int n = 0; n < 4; n++)
            bv[n] = load_bf16x16(vb + (size_t)(16 * n + li) * T_ + c0 + hi * 16);

        const bool need_mask = (c0 + 31 > m0);
#pragma unroll
        for (int r = 0; r < 8; r++) {
            const int qrow = m0 + r + 8 * hi;
            float s0 = s[0][r] * 0.125f;                    // scale = 64^-0.5
            float s1 = s[1][r] * 0.125f;
            if (need_mask) {
                if (c0 + li > qrow)      s0 = -1e30f;
                if (c0 + 16 + li > qrow) s1 = -1e30f;
            }
            float mx    = rmax16(fmaxf(s0, s1));
            float mnew  = fmaxf(mrow[r], mx);
            float alpha = __expf(mrow[r] - mnew);
            float p0 = __expf(s0 - mnew);
            float p1 = __expf(s1 - mnew);
            lrow[r] = lrow[r] * alpha + rsum16(p0 + p1);
            mrow[r] = mnew;
#pragma unroll
            for (int n = 0; n < 4; n++) o[n][r] *= alpha;
            const int row = r + 8 * hi;
            pbuf[row * 32 + li]      = (__bf16)p0;
            pbuf[row * 32 + 16 + li] = (__bf16)p1;
        }
        // re-read P in A-matrix layout (DS ops are in-order within a wave)
        bfx16 ap;
        {
            const __bf16* pp = &pbuf[li * 32 + hi * 16];
            bfx8 c0v = *(const bfx8*)pp;
            bfx8 c1v = *(const bfx8*)(pp + 8);
#pragma unroll
            for (int i = 0; i < 8; i++) { ap[i] = c0v[i]; ap[8 + i] = c1v[i]; }
        }
#pragma unroll
        for (int n = 0; n < 4; n++)
            o[n] = __builtin_amdgcn_wmma_f32_16x16x32_bf16(false, ap, false, bv[n], (short)0, o[n], false, false);
    }

#pragma unroll
    for (int n = 0; n < 4; n++) {
        const int h = 16 * n + li;
#pragma unroll
        for (int r = 0; r < 8; r++) {
            const int t = m0 + r + 8 * hi;
            out[((size_t)b * T_ + t) * HS + h] = o[n][r] / lrow[r];
        }
    }
}

// ---------------------------------------------------------------------------------------
extern "C" void kernel_launch(void* const* d_in, const int* in_sizes, int n_in,
                              void* d_out, int out_size, void* d_ws, size_t ws_size,
                              hipStream_t stream) {
    const float* x  = (const float*)d_in[0];
    const float* Wq = (const float*)d_in[1];
    const float* Wk = (const float*)d_in[2];
    const float* Wv = (const float*)d_in[3];

    float* out   = (float*)d_out;                       // [B,T,64]
    float* out_k = out   + (size_t)B_ * T_ * HS;        // [B,T,64]
    float* out_v = out_k + (size_t)B_ * T_ * HS;        // [B,T,64]

    char* ws = (char*)d_ws;
    __bf16* wbf  = (__bf16*)(ws);                       // 3*64*1024 bf16  = 384 KB
    __bf16* qbf  = (__bf16*)(ws + 393216);              // B*T*64 bf16    = 2 MB
    __bf16* kbf  = (__bf16*)(ws + 2490368);             // B*T*64 bf16    = 2 MB
    __bf16* vTbf = (__bf16*)(ws + 4587520);             // B*64*T bf16    = 2 MB

    convert_w_kernel<<<(HS * C_) / 256, 256, 0, stream>>>(Wq, Wk, Wv, wbf);
    qkv_rope_kernel<<<(B_ * T_ / 16) / 4, 128, 0, stream>>>(x, wbf, out_k, out_v, qbf, kbf, vTbf);
    attn_kernel<<<dim3(T_ / 16, B_), 32, 0, stream>>>(qbf, kbf, vTbf, out);
}
